// cuBladeW8A16LinearLayer_61753039782356
// MI455X (gfx1250) — compile-verified, matrix operand order checked by reading, not run on hardware
//
#include <hip/hip_runtime.h>

typedef __attribute__((ext_vector_type(16))) _Float16 v16h;
typedef __attribute__((ext_vector_type(8)))  _Float16 v8h;
typedef __attribute__((ext_vector_type(4)))  _Float16 v4h;
typedef __attribute__((ext_vector_type(8)))  float    v8f;
typedef __attribute__((ext_vector_type(4)))  int      v4i;

#define M_DIM 64
#define K_DIM 8192
#define N_DIM 32768
#define KC    2048            // K-chunk staged in LDS per outer iteration
#define ROWP  (KC + 8)        // padded LDS row stride in halves (4112 B -> rows 4 banks apart)
#define LDS_BYTES (M_DIM * ROWP * 2)   // 263,168 B (< 320 KB/WG on CDNA5)
#define KMASK (K_DIM - 1)     // K is a power of two
#define PFDIST 1024           // prefetch distance in int32 elems (4 KB, ~32 K-steps ahead)

// ---------------------------------------------------------------------------
// Pre-pass: x fp32 [64 x 8192] -> f16 row-major in workspace (1 MB, L2-resident)
// ---------------------------------------------------------------------------
__global__ void xcvt_kernel(const float* __restrict__ x, _Float16* __restrict__ xh) {
    const int i = (blockIdx.x * blockDim.x + threadIdx.x) * 4;
    const float4 v = *(const float4*)(x + i);
    v4h o;
    o[0] = (_Float16)v.x;
    o[1] = (_Float16)v.y;
    o[2] = (_Float16)v.z;
    o[3] = (_Float16)v.w;
    *(v4h*)(xh + i) = o;
}

// ---------------------------------------------------------------------------
// W8A16 GEMM via V_WMMA_F32_16X16X32_F16.
//   block = 256 thr (8 waves), block tile = 64(M) x 256(N), grid = 128
//   wave tile = 64(M) x 32(N) -> 4 M-tiles x 2 N-tiles, 8 v8f accumulators
//   A: staged in LDS in 64x2048 f16 chunks; inner loop is barrier-free
//   B: int8-in-int32 weights. Latency hiding is two-level:
//      (1) global_prefetch_b8 stream 4 KB ahead covers HBM latency at zero
//          VGPR / LOADcnt cost;
//      (2) 2 rotating 32-dword raw buffers (in-place reload, no copies)
//          cover L2-hit latency with ~8-16 b128 NT loads in flight.
//   Total VGPRs kept < 256 (no s_set_vgpr_msb churn, safe occupancy).
// ---------------------------------------------------------------------------
__global__ __launch_bounds__(256)
void w8a16_wmma_kernel(const _Float16* __restrict__ xh,
                       const int*      __restrict__ wq,
                       const float*    __restrict__ scales,
                       const float*    __restrict__ bias,
                       float*          __restrict__ out) {
    extern __shared__ _Float16 As[];

    const int tid  = threadIdx.x;
    const int wave = tid >> 5;
    const int lane = tid & 31;
    const int l16  = lane & 15;
    const int sel  = lane >> 4;       // 0: K 0..7 & 16..23 ; 1: K 8..15 & 24..31
    const int sel8 = sel * 8;

    const int nbase = blockIdx.x * 256 + wave * 32;
    const int ncol0 = nbase + l16;
    const int ncol1 = ncol0 + 16;

    const int* __restrict__ wrow0 = wq + (long)ncol0 * K_DIM;
    const int* __restrict__ wrow1 = wq + (long)ncol1 * K_DIM;

    // A staging assignment: 4 threads per row, 8-half segments
    const int arow = tid >> 2;
    const int acol = (tid & 3) * 8;

    v8f acc[4][2];
    #pragma unroll
    for (int m = 0; m < 4; ++m)
        #pragma unroll
        for (int j = 0; j < 2; ++j)
            acc[m][j] = (v8f){0.f, 0.f, 0.f, 0.f, 0.f, 0.f, 0.f, 0.f};

    // --- raw-weight fetch: 8 x b128 NT loads for one K=32 step (2 N-tiles) ---
    auto loadb = [&](v4i* dst, int gk) {
        const int* q0 = wrow0 + gk + sel8;
        const int* q1 = wrow1 + gk + sel8;
        dst[0] = __builtin_nontemporal_load((const v4i*)(q0));
        dst[1] = __builtin_nontemporal_load((const v4i*)(q0 + 4));
        dst[2] = __builtin_nontemporal_load((const v4i*)(q0 + 16));
        dst[3] = __builtin_nontemporal_load((const v4i*)(q0 + 20));
        dst[4] = __builtin_nontemporal_load((const v4i*)(q1));
        dst[5] = __builtin_nontemporal_load((const v4i*)(q1 + 4));
        dst[6] = __builtin_nontemporal_load((const v4i*)(q1 + 16));
        dst[7] = __builtin_nontemporal_load((const v4i*)(q1 + 20));
    };

    // --- one K=32 step: dequant B, read A frags from LDS, 8 WMMAs ---
    auto compute = [&](const v4i* br, int kk) {
        v16h bf[2];
        #pragma unroll
        for (int j = 0; j < 2; ++j) {
            v16h b;
            #pragma unroll
            for (int i = 0; i < 4; ++i) {
                b[i]      = (_Float16)(short)br[4 * j + 0][i];
                b[4 + i]  = (_Float16)(short)br[4 * j + 1][i];
                b[8 + i]  = (_Float16)(short)br[4 * j + 2][i];
                b[12 + i] = (_Float16)(short)br[4 * j + 3][i];
            }
            bf[j] = b;
        }
        #pragma unroll
        for (int m = 0; m < 4; ++m) {
            const int mrow = m * 16 + l16;
            const v8h alo = *(const v8h*)(&As[mrow * ROWP + kk + sel8]);
            const v8h ahi = *(const v8h*)(&As[mrow * ROWP + kk + 16 + sel8]);
            v16h a;
            #pragma unroll
            for (int i = 0; i < 8; ++i) { a[i] = alo[i]; a[8 + i] = ahi[i]; }
            acc[m][0] = __builtin_amdgcn_wmma_f32_16x16x32_f16(
                false, a, false, bf[0], (short)0, acc[m][0], false, false);
            acc[m][1] = __builtin_amdgcn_wmma_f32_16x16x32_f16(
                false, a, false, bf[1], (short)0, acc[m][1], false, false);
        }
    };

    // prime 2 rotating raw-B buffers with k = 0, 32
    v4i B[2][8];
    loadb(B[0], 0);
    loadb(B[1], 32);

    for (int c = 0; c < K_DIM / KC; ++c) {
        const int kbase = c * KC;

        // ---- stage 64 x KC A-chunk into LDS ----
        __syncthreads();
        #pragma unroll 4
        for (int j = 0; j < KC / 32; ++j) {
            const int col = acol + j * 32;
            const v8h av = *(const v8h*)(xh + arow * K_DIM + kbase + col);
            *(v8h*)(&As[arow * ROWP + col]) = av;
        }
        __syncthreads();

        // ---- barrier-free inner loop: 2 K-steps / iteration, in-place rotation ----
        for (int kk = 0; kk < KC; kk += 64) {
            #pragma unroll
            for (int u = 0; u < 2; ++u) {
                const int g   = (kbase + kk + 64 + u * 32) & KMASK;   // reload target
                const int gpf = (g + PFDIST) & KMASK;

                // lead with the prefetch stream (global_prefetch_b8, no counters)
                __builtin_prefetch(wrow0 + gpf + sel8, 0, 1);
                __builtin_prefetch(wrow1 + gpf + sel8, 0, 1);

                compute(B[u], kk + u * 32);
                loadb(B[u], g);            // in-place reload for k + 64
            }
        }
    }

    // ---- epilogue: per-channel scale + bias ----
    const float s0  = scales[ncol0];
    const float s1  = scales[ncol1];
    const float bb0 = bias[ncol0];
    const float bb1 = bias[ncol1];

    #pragma unroll
    for (int m = 0; m < 4; ++m) {
        #pragma unroll
        for (int r = 0; r < 8; ++r) {
            const int row = m * 16 + sel * 8 + r;   // C/D layout: VGPR r, lane half
            out[(long)row * N_DIM + ncol0] = acc[m][0][r] * s0 + bb0;
            out[(long)row * N_DIM + ncol1] = acc[m][1][r] * s1 + bb1;
        }
    }
}

// ---------------------------------------------------------------------------
extern "C" void kernel_launch(void* const* d_in, const int* in_sizes, int n_in,
                              void* d_out, int out_size, void* d_ws, size_t ws_size,
                              hipStream_t stream) {
    const float* x      = (const float*)d_in[0];
    const int*   wq     = (const int*)d_in[1];
    const float* scales = (const float*)d_in[2];
    const float* bias   = (const float*)d_in[3];
    float*       out    = (float*)d_out;
    _Float16*    xh     = (_Float16*)d_ws;   // 1 MB scratch

    xcvt_kernel<<<(M_DIM * K_DIM) / (256 * 4), 256, 0, stream>>>(x, xh);

    w8a16_wmma_kernel<<<N_DIM / 256, 256, LDS_BYTES, stream>>>(xh, wq, scales, bias, out);
}
